// tRNTN_75849122447514
// MI455X (gfx1250) — matrix operand order repaired and checked by reading, not compiled, not run on hardware
//
#include <hip/hip_runtime.h>
#include <hip/hip_bf16.h>

// ---------------------------------------------------------------------------
// tRNTN on MI455X (gfx1250): fused (concat | kron) x W GEMM per tree level
// using v_wmma_f32_16x16x32_f16. Kron rows are built on-the-fly in registers
// (v_pk_mul_f16 of a cached f16 operand vector by a scalar) from LDS-staged
// L/R vectors — the 1GB-per-level Kronecker tensor never touches memory.
// ---------------------------------------------------------------------------

typedef __attribute__((ext_vector_type(16))) _Float16 v16h;
typedef __attribute__((ext_vector_type(8)))  float    v8f;

#define BATCH      1024
#define TREES      2048              // left + right trees
#define DIM        64
#define KCAT       128               // concat(L,R)
#define KKRON      4096              // outer product
#define KTOT       (KCAT + KKRON)    // 4224
#define KCHUNKS    (KTOT / 32)       // 132
#define NTILES     (DIM / 16)        // 4
#define FRAG_ELEMS (KCHUNKS * NTILES * 32 * 16)   // 270336 f16 per weight set
#define WPB        4                 // waves per block (128 threads)

// ------------------------- weight packing ----------------------------------
// B fragment layout (contiguous k order used here, same as a_kk; B 32x16 f16, wave32):
//   lane L: column n = L&15 ; element e (0..15): K = 16*(L>>4) + e
// frag[((c*NTILES + nt)*32 + lane)*16 + e] = W[c*32 + K][nt*16 + n]
// where W[k][out] = w_small[out*128 + k]        (k < 128)
//                 = w_big  [out*4096 + (k-128)] (k >= 128)
__global__ void prep_frag_kernel(const float* __restrict__ w_small,
                                 const float* __restrict__ w_big,
                                 _Float16* __restrict__ frag) {
    int idx = blockIdx.x * blockDim.x + threadIdx.x;
    if (idx >= FRAG_ELEMS) return;
    int e    = idx & 15;
    int lane = (idx >> 4) & 31;
    int nt   = (idx >> 9) & 3;
    int c    = idx >> 11;
    int n    = nt * 16 + (lane & 15);
    int k    = c * 32 + ((lane >> 4) << 4) + e;
    float v  = (k < KCAT) ? w_small[n * KCAT + k]
                          : w_big[n * KKRON + (k - KCAT)];
    frag[idx] = (_Float16)v;
}

// ------------------------- embedding gather --------------------------------
// x0[tree][leaf][d] = voc_w[d][token] + voc_b[d]   (tree<1024: left, else right)
__global__ void embed_kernel(const int* __restrict__ ll, const int* __restrict__ rl,
                             const float* __restrict__ voc_w,
                             const float* __restrict__ voc_b,
                             float* __restrict__ x0) {
    int idx = blockIdx.x * blockDim.x + threadIdx.x;     // over TREES*64*64
    if (idx >= TREES * 64 * DIM) return;
    int d    = idx & 63;
    int row  = idx >> 6;           // tree*64 + leaf
    int tree = row >> 6;
    int leaf = row & 63;
    int tok  = (tree < BATCH) ? ll[tree * 64 + leaf]
                              : rl[(tree - BATCH) * 64 + leaf];
    x0[idx] = voc_w[(size_t)d * 50000 + tok] + voc_b[d];
}

// ------------------------- A-fragment K mapping ----------------------------
// A 16x32 f16 layout: lane: m = lane&15, half = lane>>4
//   element e -> K = half*16 + e
__device__ __forceinline__ int a_kk(int e, int half) {
    return half * 16 + e;
}

// scalar * f16-vector -> A fragment (forms v_pk_mul_f16)
__device__ __forceinline__ v16h mulh(v16h v, _Float16 s) {
    v16h r;
#pragma unroll
    for (int e = 0; e < 16; ++e) r[e] = v[e] * s;
    return r;
}

// one K-chunk: load 4 B fragments (contiguous 32B/lane) and do 4 WMMAs
__device__ __forceinline__ void wmma4(const _Float16* __restrict__ wfrag, int c,
                                      int lane, v16h a,
                                      v8f& A0, v8f& A1, v8f& A2, v8f& A3) {
    const _Float16* bf = wfrag + ((size_t)(c * NTILES) * 32 + lane) * 16;
    v16h b0 = *(const v16h*)(bf);
    v16h b1 = *(const v16h*)(bf + 512);
    v16h b2 = *(const v16h*)(bf + 1024);
    v16h b3 = *(const v16h*)(bf + 1536);
    A0 = __builtin_amdgcn_wmma_f32_16x16x32_f16(false, a, false, b0, (short)0, A0, false, false);
    A1 = __builtin_amdgcn_wmma_f32_16x16x32_f16(false, a, false, b1, (short)0, A1, false, false);
    A2 = __builtin_amdgcn_wmma_f32_16x16x32_f16(false, a, false, b2, (short)0, A2, false, false);
    A3 = __builtin_amdgcn_wmma_f32_16x16x32_f16(false, a, false, b3, (short)0, A3, false, false);
}

// ------------------------- one tree level ----------------------------------
// One wave per 16-node tile. GEMM: [16 x 4224] x [4224 x 64] with A built
// on the fly: K<64: L[k]; K<128: R[k-64]; else kron L_i * L_j.
__global__ __launch_bounds__(32 * WPB)
void compose_level_kernel(const float* __restrict__ xin,   // [T][2P][64]
                          float* __restrict__ xout,        // [T][P][64]
                          const _Float16* __restrict__ wfrag,
                          const float* __restrict__ cps_b,
                          const float* __restrict__ cpst_b,
                          int P, int pShift) {
    __shared__ float lsh[WPB][16][65];
    __shared__ float rsh[WPB][16][65];
    const int lane = threadIdx.x & 31;
    const int wave = threadIdx.x >> 5;
    const int tile = blockIdx.x * WPB + wave;
    const int totalNodes = TREES * P;
    const int base = tile * 16;
    if (base >= totalNodes) return;    // wave-uniform; active waves keep EXEC all-1s

    // Stage 16 L rows + 16 R rows (float4 segments, wave-local LDS slice).
    for (int u = lane; u < 32 * 16; u += 32) {
        int r = u >> 4, seg = u & 15;
        int i = r >> 1, isR = r & 1;
        int node = base + i;
        int t = node >> pShift, p = node & (P - 1);
        int inrow = (t << (pShift + 1)) + (p << 1) + isR;   // t*2P + 2p + isR
        float4 v = *(const float4*)(xin + (size_t)inrow * DIM + seg * 4);
        float* dst = (isR ? &rsh[wave][i][0] : &lsh[wave][i][0]) + seg * 4;
        dst[0] = v.x; dst[1] = v.y; dst[2] = v.z; dst[3] = v.w;
    }
    // Same-wave LDS ops are in-order (DScnt); no cross-wave sharing -> no barrier.

    const int m = lane & 15;
    const int half = lane >> 4;
    const float* Lrow = &lsh[wave][m][0];
    const float* Rrow = &rsh[wave][m][0];

    // Hoist the only 4 distinct operand vectors, pre-converted to f16 (v16h).
    v16h hl0, hl1, hr0, hr1;
#pragma unroll
    for (int e = 0; e < 16; ++e) {
        int kk = a_kk(e, half);
        hl0[e] = (_Float16)Lrow[kk];      hl1[e] = (_Float16)Lrow[32 + kk];
        hr0[e] = (_Float16)Rrow[kk];      hr1[e] = (_Float16)Rrow[32 + kk];
    }

    v8f acc0 = {}, acc1 = {}, acc2 = {}, acc3 = {};
    // concat chunks: K = [L(0..63) | R(0..63)]
    wmma4(wfrag, 0, lane, hl0, acc0, acc1, acc2, acc3);
    wmma4(wfrag, 1, lane, hl1, acc0, acc1, acc2, acc3);
    wmma4(wfrag, 2, lane, hr0, acc0, acc1, acc2, acc3);
    wmma4(wfrag, 3, lane, hr1, acc0, acc1, acc2, acc3);
    // kron chunks: chunk 4+2i covers kron[i*64 + 0..31] = L_i * L_j
#pragma unroll 2
    for (int i = 0; i < DIM; ++i) {
        _Float16 hli = (_Float16)Lrow[i];
        // prefetch B stream ~8 chunks (32 KB) ahead into caches
        __builtin_prefetch(wfrag + ((size_t)((12 + 2 * i) * NTILES) * 32 + lane) * 16, 0, 1);
        wmma4(wfrag, 4 + 2 * i,     lane, mulh(hl0, hli), acc0, acc1, acc2, acc3);
        wmma4(wfrag, 4 + 2 * i + 1, lane, mulh(hl1, hli), acc0, acc1, acc2, acc3);
    }

    // Epilogue: C layout -> n = lane&15 (+16*nt), M = r + 8*half. tanh + bias.
    const int n0 = lane & 15;
#pragma unroll
    for (int nt = 0; nt < NTILES; ++nt) {
        v8f av = (nt == 0) ? acc0 : (nt == 1) ? acc1 : (nt == 2) ? acc2 : acc3;
        int n = nt * 16 + n0;
        float bias = cps_b[n] + cpst_b[n];
#pragma unroll
        for (int r = 0; r < 8; ++r) {
            int node = base + half * 8 + r;     // node == t*P + p == output row
            xout[(size_t)node * DIM + n] = tanhf(av[r] + bias);
        }
    }
}

// ------------------------- final relation + softmax ------------------------
__global__ __launch_bounds__(32 * WPB)
void final_relation_kernel(const float* __restrict__ x,   // [2048][64]: l then r
                           const _Float16* __restrict__ wfrag,
                           const float* __restrict__ cpr_b,
                           const float* __restrict__ cprt_b,
                           const float* __restrict__ sm_w,
                           const float* __restrict__ sm_b,
                           float* __restrict__ out) {
    __shared__ float lsh[WPB][16][65];
    __shared__ float rsh[WPB][16][65];
    __shared__ float ash[WPB][16][65];
    const int lane = threadIdx.x & 31;
    const int wave = threadIdx.x >> 5;
    const int tile = blockIdx.x * WPB + wave;     // 64 tiles total, exact grid
    const int base = tile * 16;

    for (int u = lane; u < 32 * 16; u += 32) {
        int r = u >> 4, seg = u & 15;
        int i = r >> 1, isR = r & 1;
        int row = (isR ? BATCH : 0) + base + i;
        float4 v = *(const float4*)(x + (size_t)row * DIM + seg * 4);
        float* dst = (isR ? &rsh[wave][i][0] : &lsh[wave][i][0]) + seg * 4;
        dst[0] = v.x; dst[1] = v.y; dst[2] = v.z; dst[3] = v.w;
    }

    const int m = lane & 15;
    const int half = lane >> 4;
    const float* Lrow = &lsh[wave][m][0];
    const float* Rrow = &rsh[wave][m][0];

    v16h hl0, hl1, hr0, hr1;
#pragma unroll
    for (int e = 0; e < 16; ++e) {
        int kk = a_kk(e, half);
        hl0[e] = (_Float16)Lrow[kk];      hl1[e] = (_Float16)Lrow[32 + kk];
        hr0[e] = (_Float16)Rrow[kk];      hr1[e] = (_Float16)Rrow[32 + kk];
    }

    v8f acc0 = {}, acc1 = {}, acc2 = {}, acc3 = {};
    wmma4(wfrag, 0, lane, hl0, acc0, acc1, acc2, acc3);
    wmma4(wfrag, 1, lane, hl1, acc0, acc1, acc2, acc3);
    wmma4(wfrag, 2, lane, hr0, acc0, acc1, acc2, acc3);
    wmma4(wfrag, 3, lane, hr1, acc0, acc1, acc2, acc3);
    // kron = l_i * r_j
#pragma unroll 2
    for (int i = 0; i < DIM; ++i) {
        _Float16 hli = (_Float16)Lrow[i];
        __builtin_prefetch(wfrag + ((size_t)((12 + 2 * i) * NTILES) * 32 + lane) * 16, 0, 1);
        wmma4(wfrag, 4 + 2 * i,     lane, mulh(hr0, hli), acc0, acc1, acc2, acc3);
        wmma4(wfrag, 4 + 2 * i + 1, lane, mulh(hr1, hli), acc0, acc1, acc2, acc3);
    }

    // leaky_relu epilogue -> LDS staging for the tiny 7x64 softmax head
    const int n0 = lane & 15;
#pragma unroll
    for (int nt = 0; nt < NTILES; ++nt) {
        v8f av = (nt == 0) ? acc0 : (nt == 1) ? acc1 : (nt == 2) ? acc2 : acc3;
        int n = nt * 16 + n0;
        float bias = cpr_b[n] + cprt_b[n];
#pragma unroll
        for (int r = 0; r < 8; ++r) {
            float v = av[r] + bias;
            ash[wave][half * 8 + r][n] = (v > 0.0f) ? v : 0.01f * v;
        }
    }
    __syncthreads();   // all waves active (exact grid), safe

    if (lane < 16) {
        int b = base + lane;
        float lg[7], mx = -1e30f;
#pragma unroll
        for (int rel = 0; rel < 7; ++rel) {
            float s = sm_b[rel];
            for (int e = 0; e < DIM; ++e)
                s += ash[wave][lane][e] * sm_w[rel * DIM + e];
            lg[rel] = s;
            mx = fmaxf(mx, s);
        }
        float sum = 0.0f;
#pragma unroll
        for (int rel = 0; rel < 7; ++rel) { lg[rel] = expf(lg[rel] - mx); sum += lg[rel]; }
        float inv = 1.0f / sum;
#pragma unroll
        for (int rel = 0; rel < 7; ++rel) out[b * 7 + rel] = lg[rel] * inv;
    }
}

// ------------------------- host orchestration ------------------------------
extern "C" void kernel_launch(void* const* d_in, const int* in_sizes, int n_in,
                              void* d_out, int out_size, void* d_ws, size_t ws_size,
                              hipStream_t stream) {
    const int*   ll     = (const int*)d_in[0];
    const int*   rl     = (const int*)d_in[1];
    const float* voc_w  = (const float*)d_in[2];
    const float* voc_b  = (const float*)d_in[3];
    const float* cps_w  = (const float*)d_in[4];
    const float* cps_b  = (const float*)d_in[5];
    const float* cpst_w = (const float*)d_in[6];
    const float* cpst_b = (const float*)d_in[7];
    const float* cpr_w  = (const float*)d_in[8];
    const float* cpr_b  = (const float*)d_in[9];
    const float* cprt_w = (const float*)d_in[10];
    const float* cprt_b = (const float*)d_in[11];
    const float* sm_w   = (const float*)d_in[12];
    const float* sm_b   = (const float*)d_in[13];

    // Workspace layout (~51.4 MB):
    //   xA: [2048][64][64] f32   (33.5 MB, ping buffer; holds final l/r too)
    //   xB: [2048][32][64] f32   (16.8 MB, pong buffer)
    //   w1f, w2f: packed f16 B-fragments (528 KB each)
    char* ws = (char*)d_ws;
    float*    xA  = (float*)ws;
    float*    xB  = (float*)(ws + (size_t)TREES * 64 * DIM * 4);
    _Float16* w1f = (_Float16*)(ws + (size_t)TREES * 64 * DIM * 4
                                   + (size_t)TREES * 32 * DIM * 4);
    _Float16* w2f = w1f + FRAG_ELEMS;
    float* bufs[2] = {xA, xB};

    prep_frag_kernel<<<(FRAG_ELEMS + 255) / 256, 256, 0, stream>>>(cps_w, cpst_w, w1f);
    prep_frag_kernel<<<(FRAG_ELEMS + 255) / 256, 256, 0, stream>>>(cpr_w, cprt_w, w2f);

    int embN = TREES * 64 * DIM;
    embed_kernel<<<(embN + 255) / 256, 256, 0, stream>>>(ll, rl, voc_w, voc_b, xA);

    int cur = 0;
    for (int P = 32; P >= 1; P >>= 1) {       // 6 levels: 64->32->...->1 nodes
        int tiles  = (TREES * P) / 16;        // = 128*P, always multiple of WPB
        int blocks = tiles / WPB;
        compose_level_kernel<<<blocks, 32 * WPB, 0, stream>>>(
            bufs[cur], bufs[cur ^ 1], w1f, cps_b, cpst_b, P, __builtin_ctz(P));
        cur ^= 1;
    }
    // 6 flips -> result back in xA, rows [0,1024)=l, [1024,2048)=r

    final_relation_kernel<<<(BATCH / 16) / WPB, 32 * WPB, 0, stream>>>(
        xA, w2f, cpr_b, cprt_b, sm_w, sm_b, (float*)d_out);
}
